// NCell_22273700397573
// MI455X (gfx1250) — matrix operand order repaired and checked by reading, not compiled, not run on hardware
//
#include <hip/hip_runtime.h>
#include <hip/hip_bf16.h>

typedef __attribute__((ext_vector_type(16))) __bf16 v16bf;
typedef __attribute__((ext_vector_type(8)))  __bf16 v8bf;
typedef __attribute__((ext_vector_type(8)))  float  v8f;

#define Bsz 64
#define Cc 128
#define Hh 32
#define Ww 32
#define HW 1024
#define CHW (Cc*HW)        // 131072
#define OUT_BS (5*CHW)     // 655360 (b-stride of concat output)
#define NT_PIX 128         // pixels per workgroup tile (4 image rows)
#define LR 8               // halo rows (4 + 2+2)
#define LC 36              // halo cols (32 + 2+2)
#define LCI 32             // ci chunk resident in LDS
#define LCP 40             // padded ci stride: 80B = 20 banks -> conflict-free b128
#define LDS_REAL (LR*LC*LCI)

// ---------------------------------------------------------------------------
// Weight prep: softmax(alpha) scaling folded into bf16 weights, repacked
// OIHW -> [tap][co][ci] so WMMA A-fragments are contiguous.
// ---------------------------------------------------------------------------
__global__ void prep_weights_kernel(const float* __restrict__ wc, const float* __restrict__ wd,
                                    const float* __restrict__ wp0, const float* __restrict__ wp1,
                                    const float* __restrict__ alphas,
                                    __bf16* __restrict__ wcb, __bf16* __restrict__ wdb,
                                    __bf16* __restrict__ wp0b, __bf16* __restrict__ wp1b) {
  long idx = (long)blockIdx.x * 256 + threadIdx.x;
  const long NCV = 8L * 9 * Cc * Cc;
  if (idx < 2 * NCV) {
    int isd = idx >= NCV;
    long r = isd ? idx - NCV : idx;
    int j   = (int)(r / (9 * Cc * Cc));
    int rem = (int)(r % (9 * Cc * Cc));
    int tap = rem / (Cc * Cc);
    int rr  = rem % (Cc * Cc);
    int co = rr / Cc, ci = rr % Cc;
    float a[5]; float m = -1e30f;
    #pragma unroll
    for (int k = 0; k < 5; k++) { a[k] = alphas[j * 5 + k]; m = fmaxf(m, a[k]); }
    float den = 0.f;
    #pragma unroll
    for (int k = 0; k < 5; k++) { a[k] = __expf(a[k] - m); den += a[k]; }
    float scale = (isd ? a[2] : a[1]) / den;
    const float* src = isd ? wd : wc;
    float v = src[((long)(j * Cc + co) * Cc + ci) * 9 + tap] * scale;
    __bf16* dst = isd ? wdb : wcb;
    dst[((long)(j * 9 + tap) * Cc + co) * Cc + ci] = (__bf16)v;
  } else {
    long r = idx - 2 * NCV;
    if (r < Cc * Cc)            wp0b[r] = (__bf16)wp0[r];
    else if (r < 2L * Cc * Cc)  wp1b[r - Cc * Cc] = (__bf16)wp1[r - Cc * Cc];
  }
}

// ---------------------------------------------------------------------------
// 1x1 preprocess conv: s[b,co,hw] = sum_ci wpre[co,ci] * in[b,ci,hw]
// Wave tiling: 32 co x 64 pix (2 co-tiles x 4 n-tiles), B fragment reused
// across the two co-tiles from registers.
// ---------------------------------------------------------------------------
__global__ void pre_kernel(const float* __restrict__ in, const __bf16* __restrict__ w,
                           float* __restrict__ out) {
  __shared__ __align__(32) __bf16 lds[NT_PIX * LCP]; // 10 KB
  const int tile = blockIdx.x, b = blockIdx.y;
  const int tid = threadIdx.x;
  const int lane = tid & 31, wv = tid >> 5;
  const int l15 = lane & 15;
  const int hi = (lane >> 4) & 1;
  const int cg = wv & 3, pg = wv >> 2;
  const int cobase = cg * 32;
  const int pixbase = pg * 64;
  v8f zero = {0.f,0.f,0.f,0.f,0.f,0.f,0.f,0.f};
  v8f acc[2][4];
  #pragma unroll
  for (int ct = 0; ct < 2; ct++)
    #pragma unroll
    for (int nt = 0; nt < 4; nt++) acc[ct][nt] = zero;

  const float* inb = in + (long)b * CHW + tile * NT_PIX;
  for (int kc = 0; kc < 4; kc++) {
    if (kc) __syncthreads();
    for (int i = tid; i < NT_PIX * LCI; i += 256) {
      int ci = i >> 7, p = i & 127;
      lds[p * LCP + ci] = (__bf16)inb[(ci + kc * LCI) * HW + p];
    }
    __syncthreads();
    union { v16bf v; v8bf h[2]; } ua[2];
    #pragma unroll
    for (int ct = 0; ct < 2; ct++) {
      const __bf16* wrow = w + (cobase + ct * 16 + l15) * Cc + kc * LCI + hi * 8;
      ua[ct].h[0] = *(const v8bf*)(wrow);
      ua[ct].h[1] = *(const v8bf*)(wrow + 16);
    }
    const __bf16* lb = lds + hi * 8;
    #pragma unroll
    for (int nt = 0; nt < 4; nt++) {
      int pix = pixbase + nt * 16 + l15;
      const v8bf* bp = (const v8bf*)(lb + pix * LCP);
      union { v16bf v; v8bf h[2]; } ub;
      ub.h[0] = bp[0]; ub.h[1] = bp[2];
      acc[0][nt] = __builtin_amdgcn_wmma_f32_16x16x32_bf16(
          false, ua[0].v, false, ub.v, (short)0, acc[0][nt], false, false);
      acc[1][nt] = __builtin_amdgcn_wmma_f32_16x16x32_bf16(
          false, ua[1].v, false, ub.v, (short)0, acc[1][nt], false, false);
    }
  }
  float* ob = out + (long)b * CHW + tile * NT_PIX;
  #pragma unroll
  for (int ct = 0; ct < 2; ct++)
    #pragma unroll
    for (int nt = 0; nt < 4; nt++) {
      int pix = pixbase + nt * 16 + l15;
      #pragma unroll
      for (int r = 0; r < 8; r++)
        ob[(cobase + ct * 16 + r + hi * 8) * HW + pix] = acc[ct][nt][r];
    }
}

// ---------------------------------------------------------------------------
// Node kernel: out = mixed_op(h0,...) + mixed_op(h1,...).
// Conv + dilated conv via 9 shifted K=128 bf16 WMMA GEMMs from an LDS halo
// tile (padded stride -> conflict-free ds_load_b128); identity/avg/max fused
// from the same tile when the wave's 32-co group matches the resident chunk.
// ---------------------------------------------------------------------------
__global__ void node_kernel(const float* __restrict__ h0, long h0bs,
                            const float* __restrict__ h1, long h1bs,
                            float* __restrict__ out,
                            const __bf16* __restrict__ wAc, const __bf16* __restrict__ wAd,
                            const __bf16* __restrict__ wBc, const __bf16* __restrict__ wBd,
                            const float* __restrict__ alphas, int eA, int eB) {
  __shared__ __align__(32) __bf16 lds[2][LR * LC * LCP]; // 45 KB
  const int tile = blockIdx.x, b = blockIdx.y;
  const int tid = threadIdx.x;
  const int lane = tid & 31, wv = tid >> 5;
  const int l15 = lane & 15;
  const int hi = (lane >> 4) & 1;
  const int cg = wv & 3, pg = wv >> 2;  // 32-co group == ci chunk; 64-pix group
  const int cobase = cg * 32;
  const int pixbase = pg * 64;
  const int r0 = tile * 4;

  float aA[5], aB[5];
  {
    float m = -1e30f;
    #pragma unroll
    for (int k = 0; k < 5; k++) { aA[k] = alphas[eA * 5 + k]; m = fmaxf(m, aA[k]); }
    float s = 0.f;
    #pragma unroll
    for (int k = 0; k < 5; k++) { aA[k] = __expf(aA[k] - m); s += aA[k]; }
    float inv = 1.f / s;
    #pragma unroll
    for (int k = 0; k < 5; k++) aA[k] *= inv;
  }
  {
    float m = -1e30f;
    #pragma unroll
    for (int k = 0; k < 5; k++) { aB[k] = alphas[eB * 5 + k]; m = fmaxf(m, aB[k]); }
    float s = 0.f;
    #pragma unroll
    for (int k = 0; k < 5; k++) { aB[k] = __expf(aB[k] - m); s += aB[k]; }
    float inv = 1.f / s;
    #pragma unroll
    for (int k = 0; k < 5; k++) aB[k] *= inv;
  }

  v8f zero = {0.f,0.f,0.f,0.f,0.f,0.f,0.f,0.f};
  v8f acc[2][4];
  #pragma unroll
  for (int ct = 0; ct < 2; ct++)
    #pragma unroll
    for (int nt = 0; nt < 4; nt++) acc[ct][nt] = zero;

  const float* hb0 = h0 + (long)b * h0bs;
  const float* hb1 = h1 + (long)b * h1bs;

  for (int kc = 0; kc < 4; kc++) {
    if (kc) __syncthreads();
    // Fill zero-padded halo tiles (bf16, pixel-major, padded ci stride)
    for (int i = tid; i < LDS_REAL; i += 256) {
      int ci = i / (LR * LC);
      int p  = i % (LR * LC);
      int row = p / LC, col = p % LC;
      int gr = r0 + row - 2, gc = col - 2;
      float v0 = 0.f, v1 = 0.f;
      if ((unsigned)gr < 32u && (unsigned)gc < 32u) {
        long gi = (long)(ci + kc * LCI) * HW + gr * Ww + gc;
        v0 = hb0[gi]; v1 = hb1[gi];
      }
      lds[0][p * LCP + ci] = (__bf16)v0;
      lds[1][p * LCP + ci] = (__bf16)v1;
    }
    __syncthreads();

    // identity + avg-pool + max-pool when this chunk holds our co group
    if (cg == kc) {
      for (int s2 = 0; s2 < 2; s2++) {
        float c0 = s2 ? aB[0] : aA[0];
        float c3 = s2 ? aB[3] : aA[3];
        float c4 = s2 ? aB[4] : aA[4];
        #pragma unroll
        for (int ct = 0; ct < 2; ct++) {
          int cib = ct * 16 + hi * 8;
          #pragma unroll
          for (int nt = 0; nt < 4; nt++) {
            int pix = pixbase + nt * 16 + l15;
            int prow = pix >> 5, pcol = pix & 31;
            float sum[8], mx[8], cen[8];
            #pragma unroll
            for (int r = 0; r < 8; r++) { sum[r] = 0.f; mx[r] = -1e30f; cen[r] = 0.f; }
            #pragma unroll
            for (int dy = -1; dy <= 1; dy++) {
              #pragma unroll
              for (int dx = -1; dx <= 1; dx++) {
                const v8bf vv =
                    *(const v8bf*)&lds[s2][((prow + 2 + dy) * LC + (pcol + 2 + dx)) * LCP + cib];
                bool valid = ((unsigned)(r0 + prow + dy) < 32u) && ((unsigned)(pcol + dx) < 32u);
                #pragma unroll
                for (int r = 0; r < 8; r++) {
                  float f = (float)vv[r];
                  sum[r] += f;                        // zero halo => count_include_pad avg
                  if (valid) mx[r] = fmaxf(mx[r], f); // -inf-padded max semantics
                  if (dy == 0 && dx == 0) cen[r] = f;
                }
              }
            }
            #pragma unroll
            for (int r = 0; r < 8; r++)
              acc[ct][nt][r] += c0 * cen[r] + c3 * (sum[r] * (1.f / 9.f)) + c4 * mx[r];
          }
        }
      }
    }

    // 4 conv weight sets x 9 shifted taps, K-chunk = 32
    for (int s2 = 0; s2 < 2; s2++) {
      for (int cd = 0; cd < 2; cd++) {
        const __bf16* wb = (s2 == 0) ? (cd == 0 ? wAc : wAd) : (cd == 0 ? wBc : wBd);
        int df = (cd == 0) ? 1 : 2;
        for (int tap = 0; tap < 9; tap++) {
          int dy = (tap / 3 - 1) * df, dx = (tap % 3 - 1) * df;
          union { v16bf v; v8bf h[2]; } ua[2];
          #pragma unroll
          for (int ct = 0; ct < 2; ct++) {
            const __bf16* wrow =
                wb + ((long)(tap * Cc + cobase + ct * 16 + l15)) * Cc + kc * LCI + hi * 8;
            ua[ct].h[0] = *(const v8bf*)(wrow);
            ua[ct].h[1] = *(const v8bf*)(wrow + 16);
          }
          const __bf16* lb = &lds[s2][0] + hi * 8;
          #pragma unroll
          for (int nt = 0; nt < 4; nt++) {
            int pix = pixbase + nt * 16 + l15;
            int row = (pix >> 5) + 2 + dy, col = (pix & 31) + 2 + dx;
            const v8bf* bp = (const v8bf*)(lb + (row * LC + col) * LCP);
            union { v16bf v; v8bf h[2]; } ub;
            ub.h[0] = bp[0]; ub.h[1] = bp[2];
            acc[0][nt] = __builtin_amdgcn_wmma_f32_16x16x32_bf16(
                false, ua[0].v, false, ub.v, (short)0, acc[0][nt], false, false);
            acc[1][nt] = __builtin_amdgcn_wmma_f32_16x16x32_bf16(
                false, ua[1].v, false, ub.v, (short)0, acc[1][nt], false, false);
          }
        }
      }
    }
  }

  float* ob = out + (long)b * OUT_BS + tile * NT_PIX;
  #pragma unroll
  for (int ct = 0; ct < 2; ct++)
    #pragma unroll
    for (int nt = 0; nt < 4; nt++) {
      int pix = pixbase + nt * 16 + l15;
      #pragma unroll
      for (int r = 0; r < 8; r++)
        ob[(cobase + ct * 16 + r + hi * 8) * HW + pix] = acc[ct][nt][r];
    }
}

// ---------------------------------------------------------------------------
// skip_input -> concat chunk 4 and the second tuple output
// ---------------------------------------------------------------------------
__global__ void skip_kernel(const float4* __restrict__ skip, float* __restrict__ out) {
  long idx = (long)blockIdx.x * 256 + threadIdx.x;
  const long N = (long)Bsz * CHW / 4;
  if (idx >= N) return;
  float4 v = skip[idx];
  long e = idx * 4;
  long bb = e / CHW, r = e % CHW;
  *(float4*)(out + bb * OUT_BS + 4L * CHW + r) = v;
  *(float4*)(out + (long)Bsz * OUT_BS + e) = v;
}

extern "C" void kernel_launch(void* const* d_in, const int* in_sizes, int n_in,
                              void* d_out, int out_size, void* d_ws, size_t ws_size,
                              hipStream_t stream) {
  const float* input0  = (const float*)d_in[0];
  const float* input1  = (const float*)d_in[1];
  const float* skip_in = (const float*)d_in[2];
  const float* w_pre0  = (const float*)d_in[3];
  const float* w_pre1  = (const float*)d_in[4];
  const float* w_conv3 = (const float*)d_in[5];
  const float* w_dil3  = (const float*)d_in[6];
  const float* alphas  = (const float*)d_in[7];
  float* out = (float*)d_out;

  char* ws = (char*)d_ws;
  float*  s0   = (float*)ws;                             // 33.55 MB
  float*  s1   = (float*)(ws + (size_t)Bsz * CHW * 4);   // 33.55 MB
  __bf16* wp0b = (__bf16*)(ws + 2 * (size_t)Bsz * CHW * 4);
  __bf16* wp1b = wp0b + Cc * Cc;
  __bf16* wcb  = wp1b + Cc * Cc;                         // 8 * 9*128*128 bf16
  __bf16* wdb  = wcb + 8L * 9 * Cc * Cc;

  // 1. weight prep (softmax-alpha folded, repacked, bf16)
  long NCV = 8L * 9 * Cc * Cc;
  long tot = 2 * NCV + 2L * Cc * Cc;
  prep_weights_kernel<<<(int)((tot + 255) / 256), 256, 0, stream>>>(
      w_conv3, w_dil3, w_pre0, w_pre1, alphas, wcb, wdb, wp0b, wp1b);

  // 2. preprocess 1x1 convs
  dim3 g(HW / NT_PIX, Bsz);
  pre_kernel<<<g, 256, 0, stream>>>(input0, wp0b, s0);
  pre_kernel<<<g, 256, 0, stream>>>(input1, wp1b, s1);

  // 3. four mixed-op nodes, chained through the concat output buffer
  const float* st[5]; long bs[5];
  st[0] = s0;            bs[0] = CHW;
  st[1] = s1;            bs[1] = CHW;
  st[2] = out + 0 * CHW; bs[2] = OUT_BS;
  st[3] = out + 1 * CHW; bs[3] = OUT_BS;
  st[4] = out + 2 * CHW; bs[4] = OUT_BS;
  for (int i = 0; i < 4; i++) {
    node_kernel<<<g, 256, 0, stream>>>(
        st[i], bs[i], st[i + 1], bs[i + 1], out + (long)i * CHW,
        wcb + (2L * i)     * 9 * Cc * Cc, wdb + (2L * i)     * 9 * Cc * Cc,
        wcb + (2L * i + 1) * 9 * Cc * Cc, wdb + (2L * i + 1) * 9 * Cc * Cc,
        alphas, 2 * i, 2 * i + 1);
  }

  // 4. skip chunk + second tuple output
  long n4 = (long)Bsz * CHW / 4;
  skip_kernel<<<(int)((n4 + 255) / 256), 256, 0, stream>>>((const float4*)skip_in, out);
}
